// HyenaFilter_35905926595070
// MI455X (gfx1250) — compile-verified, matrix operand order checked by reading, not run on hardware
//
#include <hip/hip_runtime.h>
#include <hip/hip_bf16.h>
#include <math.h>

// ---------------- problem constants (match reference) ----------------
#define B_SZ     8
#define D_MODEL  1024
#define SEQ_LEN  4096
#define ORDER    64
#define FFT_N    8192          // 2*L zero-padded linear convolution
#define FFT_LOGN 13
#define KF_STRIDE 4100         // float2 stride per channel (>= N/2+1)

#define PI_F 3.14159265358979323846f
// deltas = abs(linspace(log(0.01)/1.5, log(0.01)/0.3, D))
#define MIN_D (-3.0701134573253945f)
#define MAX_D (-15.350567286626972f)

typedef __attribute__((ext_vector_type(16))) _Float16 v16h;
typedef __attribute__((ext_vector_type(8)))  float    v8f;

// ---------------------------------------------------------------------
// CDNA5 async global->LDS (ASYNCcnt-tracked DMA path), per ISA 10. / 15.18.3
// the LDS address is taken from the VDST VGPR; flat shared pointers carry the LDS
// offset in addr[31:0], so truncating the pointer gives the VDST value.
// ---------------------------------------------------------------------
__device__ __forceinline__ void async_load_b128(void* lds_dst, const void* gsrc)
{
    unsigned lds_off = (unsigned)(size_t)lds_dst;
    asm volatile("global_load_async_to_lds_b128 %0, %1, off"
                 :: "v"(lds_off), "v"(gsrc)
                 : "memory");
}
__device__ __forceinline__ void wait_asynccnt0()
{
    asm volatile("s_wait_asynccnt 0x0" ::: "memory");
}

// complex helpers
__device__ __forceinline__ float2 cmul(float2 a, float2 b)
{ return make_float2(a.x * b.x - a.y * b.y, a.x * b.y + a.y * b.x); }
__device__ __forceinline__ float2 cadd(float2 a, float2 b)
{ return make_float2(a.x + b.x, a.y + b.y); }
__device__ __forceinline__ float2 csub(float2 a, float2 b)
{ return make_float2(a.x - b.x, a.y - b.y); }

// ---------------------------------------------------------------------
// Kernel 1: tiny MLP over positional embedding -> h[L, 64] (f16)
// bands = (EMB_DIM-1)//2 = 1, f = [1e-4]
// z = [ t, cos(1e-4 * 2*pi*l/S), -sin(1e-4 * 2*pi*l/S) ]
// ---------------------------------------------------------------------
__global__ __launch_bounds__(128)
void hyena_mlp_kernel(const float* __restrict__ W1, const float* __restrict__ b1,
                      const float* __restrict__ W2, const float* __restrict__ b2,
                      const float* __restrict__ W3, const float* __restrict__ b3,
                      const float* __restrict__ freq,
                      _Float16* __restrict__ hOut)
{
    int l = blockIdx.x * blockDim.x + threadIdx.x;
    if (l >= SEQ_LEN) return;

    float t    = (float)l / (float)(SEQ_LEN - 1);
    float wang = 2.0f * PI_F * (float)l / (float)SEQ_LEN;
    float ph   = 1e-4f * wang;
    float z0 = t, z1 = cosf(ph), z2 = -sinf(ph);

    float ha[ORDER], hb[ORDER];

    #pragma unroll 4
    for (int o = 0; o < ORDER; ++o) {
        float s = z0 * W1[o] + z1 * W1[ORDER + o] + z2 * W1[2 * ORDER + o] + b1[o];
        ha[o] = sinf(freq[o] * s);
    }
    #pragma unroll 2
    for (int o = 0; o < ORDER; ++o) {
        float s = b2[o];
        for (int i = 0; i < ORDER; ++i) s += ha[i] * W2[i * ORDER + o];
        hb[o] = sinf(freq[o] * s);
    }
    #pragma unroll 2
    for (int o = 0; o < ORDER; ++o) {
        float s = b3[o];
        for (int i = 0; i < ORDER; ++i) s += hb[i] * W3[i * ORDER + o];
        ha[o] = sinf(freq[o] * s);
    }
    for (int o = 0; o < ORDER; ++o)
        hOut[(size_t)l * ORDER + o] = (_Float16)ha[o];
}

// ---------------------------------------------------------------------
// Kernel 2: WMMA projection + modulation.
// k[d,l] = (sum_k Wout[k,d] * h[l,k]) * exp(-t_l * delta_d)
// as the transposed GEMM Wout^T[1024,64] @ h^T[64,4096]; one 16x16 tile
// per wave; K=64 as two v_wmma_f32_16x16x32_f16.
// ---------------------------------------------------------------------
__global__ __launch_bounds__(256)
void hyena_project_wmma_kernel(const _Float16* __restrict__ h,    // [L, 64]
                               const float*    __restrict__ Wout, // [64, D]
                               float*          __restrict__ kOut) // [D, L]
{
    const int lane  = threadIdx.x & 31;
    const int wave  = threadIdx.x >> 5;
    const int tile  = blockIdx.x * 8 + wave;     // (D/16)*(L/16) = 16384 tiles
    const int tileD = tile & (D_MODEL / 16 - 1);
    const int tileL = tile / (D_MODEL / 16);
    const int d0 = tileD * 16;
    const int l0 = tileL * 16;

    const int hi    = lane >> 4;      // 0 | 1
    const int lo16  = lane & 15;
    const int kbase = hi * 8;         // A/B 16-bit operand K-grouping per lane

    // warm the h rows this wave needs (global_prefetch_b8)
    __builtin_prefetch(h + (size_t)(l0 + lo16) * ORDER, 0, 1);

    v8f c = {};
    #pragma unroll
    for (int kc = 0; kc < ORDER; kc += 32) {
        v16h a, b;
        // A[m][k] = Wout[(kc+k)*D + d0+m], lane row m = lo16
        #pragma unroll
        for (int j = 0; j < 8; ++j) {
            a[j]     = (_Float16)Wout[(size_t)(kc + kbase + j)      * D_MODEL + (d0 + lo16)];
            a[8 + j] = (_Float16)Wout[(size_t)(kc + kbase + 16 + j) * D_MODEL + (d0 + lo16)];
        }
        // B[k][n] = h[(l0+n)*64 + kc + k], lane column n = lo16 (contiguous)
        const _Float16* hr = h + (size_t)(l0 + lo16) * ORDER + kc;
        #pragma unroll
        for (int j = 0; j < 8; ++j) {
            b[j]     = hr[kbase + j];
            b[8 + j] = hr[kbase + 16 + j];
        }
        c = __builtin_amdgcn_wmma_f32_16x16x32_f16(
                /*neg_a=*/false, a, /*neg_b=*/false, b,
                /*c_mod=*/(short)0, c, /*reuse_a=*/false, /*reuse_b=*/false);
    }

    // Epilogue: fused exponential modulation. C/D layout: VGPR r holds
    // row (r + 8*hi), lane&15 is the column.
    const int   l  = l0 + lo16;
    const float t  = (float)l / (float)(SEQ_LEN - 1);
    #pragma unroll
    for (int r = 0; r < 8; ++r) {
        int   d     = d0 + r + 8 * hi;
        float delta = fabsf(MIN_D + (MAX_D - MIN_D) * ((float)d / (float)(D_MODEL - 1)));
        kOut[(size_t)d * SEQ_LEN + l] = c[r] * __expf(-t * delta);
    }
}

// ---------------------------------------------------------------------
// Shared-LDS DIT FFT, N=8192. Six fused stage-pairs (radix-4 dataflow:
// each thread keeps a 4-point quad in registers across two radix-2
// stages) + one final radix-2 stage. Halves LDS traffic and cuts
// barriers 13 -> 7 vs plain radix-2. Twiddle for the odd pair of the
// second stage is tw * exp(-i*pi/2) = (tw.y, -tw.x) -- no extra load.
// ---------------------------------------------------------------------
__device__ __forceinline__
void fft8192_lds(float2* __restrict__ s, const float2* __restrict__ tw, int tid)
{
    #pragma unroll 1
    for (int st = 1; st <= FFT_LOGN - 2; st += 2) {
        const int h = 1 << (st - 1);
        for (int q = tid; q < FFT_N / 4; q += 256) {
            int grp  = q >> (st - 1);
            int pos  = q & (h - 1);
            int base = (grp << (st + 1)) + pos;
            float2 e0 = s[base];
            float2 e1 = s[base + h];
            float2 e2 = s[base + 2 * h];
            float2 e3 = s[base + 3 * h];
            // stage st: (e0,e1) and (e2,e3), same twiddle
            float2 w1 = tw[pos << (FFT_LOGN - st)];
            float2 t1 = cmul(e1, w1);
            float2 t3 = cmul(e3, w1);
            float2 a0 = cadd(e0, t1), a1 = csub(e0, t1);
            float2 a2 = cadd(e2, t3), a3 = csub(e2, t3);
            // stage st+1: (a0,a2) with w2, (a1,a3) with w2*(-i)
            float2 w2 = tw[pos << (FFT_LOGN - st - 1)];
            float2 w3 = make_float2(w2.y, -w2.x);
            float2 t2 = cmul(a2, w2);
            float2 t4 = cmul(a3, w3);
            s[base]         = cadd(a0, t2);
            s[base + h]     = cadd(a1, t4);
            s[base + 2 * h] = csub(a0, t2);
            s[base + 3 * h] = csub(a1, t4);
        }
        __syncthreads();
    }
    // final radix-2 stage (st = FFT_LOGN), half = N/2, twiddle index = i
    for (int i = tid; i < FFT_N / 2; i += 256) {
        float2 w = tw[i];
        float2 u = s[i], v = s[i + FFT_N / 2];
        float2 t = cmul(v, w);
        s[i]             = cadd(u, t);
        s[i + FFT_N / 2] = csub(u, t);
    }
    __syncthreads();
}

__device__ __forceinline__ void build_twiddles(float2* tw, int tid)
{
    for (int q = tid; q < FFT_N / 2; q += 256) {
        float sn, cs;
        __sincosf(-2.0f * PI_F * (float)q / (float)FFT_N, &sn, &cs);
        tw[q] = make_float2(cs, sn);
    }
}

// ---------------------------------------------------------------------
// Kernel 3: spectra of the filters. One workgroup per channel d:
// kf[d, 0..N/2] = rfft(k[d,:], N)   (Hermitian half-spectrum)
// Row staged into LDS via async-to-LDS DMA, overlapped with twiddle build.
// ---------------------------------------------------------------------
__global__ __launch_bounds__(256)
void hyena_filter_fft_kernel(const float* __restrict__ kIn,  // [D, L]
                             float2*      __restrict__ kf)   // [D, KF_STRIDE]
{
    __shared__ float2 s[FFT_N];
    __shared__ float2 tw[FFT_N / 2];
    __shared__ __align__(16) float row[SEQ_LEN];

    const int d   = blockIdx.x;
    const int tid = threadIdx.x;
    const float* kr = kIn + (size_t)d * SEQ_LEN;

    // async DMA the 16KB filter row into LDS (ASYNCcnt path)
    for (int i = tid; i < SEQ_LEN / 4; i += 256)
        async_load_b128(&row[i * 4], kr + i * 4);

    build_twiddles(tw, tid);          // overlaps with the DMA

    wait_asynccnt0();
    __syncthreads();

    for (int j = tid; j < FFT_N; j += 256) {            // bit-reversed load + zero pad
        int rj  = __brev((unsigned)j) >> (32 - FFT_LOGN);
        float v = (j < SEQ_LEN) ? row[j] : 0.0f;
        s[rj] = make_float2(v, 0.0f);
    }
    __syncthreads();
    fft8192_lds(s, tw, tid);

    float2* kfd = kf + (size_t)d * KF_STRIDE;
    for (int j = tid; j <= FFT_N / 2; j += 256) kfd[j] = s[j];
}

// ---------------------------------------------------------------------
// Kernel 4: fused long convolution. One workgroup per (b,d) row:
// y = Re(ifft( fft(x,N) * kf )) [:L] ;  out = y + x * bias[d]
// Inverse FFT via conjugate trick reusing the forward butterflies; 1/N
// folded into the pointwise multiply. x row stays resident in LDS
// (async DMA once, reused for the residual path -> no 2nd HBM read).
// LDS: 64KB spectrum + 32KB twiddles + 16KB row = 112KB (< 320KB/WGP).
// ---------------------------------------------------------------------
__global__ __launch_bounds__(256)
void hyena_fft_conv_kernel(const float*  __restrict__ x,     // [B, D, L]
                           const float*  __restrict__ bias,  // [D]
                           const float2* __restrict__ kf,    // [D, KF_STRIDE]
                           float*        __restrict__ out)   // [B, D, L]
{
    __shared__ float2 s[FFT_N];
    __shared__ float2 tw[FFT_N / 2];
    __shared__ __align__(16) float xs[SEQ_LEN];

    const int row = blockIdx.x;                 // b*D + d
    const int d   = row & (D_MODEL - 1);
    const int tid = threadIdx.x;
    const float* xr = x   + (size_t)row * SEQ_LEN;
    float*       yr = out + (size_t)row * SEQ_LEN;

    // async DMA the 16KB x row into LDS (ASYNCcnt path)
    for (int i = tid; i < SEQ_LEN / 4; i += 256)
        async_load_b128(&xs[i * 4], xr + i * 4);

    build_twiddles(tw, tid);          // overlaps with the DMA

    wait_asynccnt0();
    __syncthreads();

    for (int j = tid; j < FFT_N; j += 256) {            // bit-reversed load + zero pad
        int rj  = __brev((unsigned)j) >> (32 - FFT_LOGN);
        float v = (j < SEQ_LEN) ? xs[j] : 0.0f;
        s[rj] = make_float2(v, 0.0f);
    }
    __syncthreads();
    fft8192_lds(s, tw, tid);                            // forward FFT

    // pointwise multiply (Hermitian reconstruction of kf), conj + 1/N
    const float2* kfd = kf + (size_t)d * KF_STRIDE;
    const float invN = 1.0f / (float)FFT_N;
    for (int j = tid; j < FFT_N; j += 256) {
        float2 X = s[j];
        float2 K;
        if (j <= FFT_N / 2) K = kfd[j];
        else { float2 kc = kfd[FFT_N - j]; K = make_float2(kc.x, -kc.y); }
        float pr = X.x * K.x - X.y * K.y;
        float pi = X.x * K.y + X.y * K.x;
        s[j] = make_float2(pr * invN, -pi * invN);      // conj: ifft = conj(fft(conj(.)))/N
    }
    __syncthreads();
    for (int j = tid; j < FFT_N; j += 256) {            // bit-reversal permutation
        int rj = __brev((unsigned)j) >> (32 - FFT_LOGN);
        if (rj > j) { float2 a2 = s[j]; s[j] = s[rj]; s[rj] = a2; }
    }
    __syncthreads();
    fft8192_lds(s, tw, tid);                            // "inverse" FFT

    const float bd = bias[d];
    for (int l = tid; l < SEQ_LEN; l += 256) {
        yr[l] = s[l].x + xs[l] * bd;                    // residual from LDS copy
    }
}

// ---------------------------------------------------------------------
// Launch
// ---------------------------------------------------------------------
extern "C" void kernel_launch(void* const* d_in, const int* in_sizes, int n_in,
                              void* d_out, int out_size, void* d_ws, size_t ws_size,
                              hipStream_t stream)
{
    const float* x    = (const float*)d_in[0];
    const float* bias = (const float*)d_in[1];
    const float* W1   = (const float*)d_in[2];
    const float* b1   = (const float*)d_in[3];
    const float* W2   = (const float*)d_in[4];
    const float* b2   = (const float*)d_in[5];
    const float* W3   = (const float*)d_in[6];
    const float* b3   = (const float*)d_in[7];
    const float* Wout = (const float*)d_in[8];
    const float* freq = (const float*)d_in[9];
    float* out = (float*)d_out;

    // workspace layout
    char* ws = (char*)d_ws;
    _Float16* hbuf = (_Float16*)ws;                                     //  512 KB
    float*    kbuf = (float*)(ws + (size_t)(512 << 10));                //   16 MB
    float2*   kf   = (float2*)(ws + (size_t)(512 << 10) + (16u << 20)); // 33.6 MB

    // 1) MLP -> h[L,64] f16
    hyena_mlp_kernel<<<SEQ_LEN / 128, 128, 0, stream>>>(W1, b1, W2, b2, W3, b3, freq, hbuf);

    // 2) WMMA projection + exp modulation -> k[D,L]
    hyena_project_wmma_kernel<<<(D_MODEL / 16) * (SEQ_LEN / 16) / 8, 256, 0, stream>>>(
        hbuf, Wout, kbuf);

    // 3) filter half-spectra kf[D, N/2+1]
    hyena_filter_fft_kernel<<<D_MODEL, 256, 0, stream>>>(kbuf, kf);

    // 4) fused FFT convolution + residual bias path
    hyena_fft_conv_kernel<<<B_SZ * D_MODEL, 256, 0, stream>>>(x, bias, kf, out);
}
